// PhysicsAttention_55886114455564
// MI455X (gfx1250) — matrix operand order repaired and checked by reading, not compiled, not run on hardware
//
#include <hip/hip_runtime.h>

typedef __attribute__((ext_vector_type(16))) _Float16 v16h;
typedef __attribute__((ext_vector_type(8)))  _Float16 v8h;
typedef __attribute__((ext_vector_type(8)))  float    v8f;

#define D_MODEL 256
#define N_HEADS 8
#define D_K 32
#define BATCH 4
#define SEQ 2048
#define NROWS (BATCH * SEQ)      // 8192
#define QKV_COLS (3 * D_MODEL)   // 768

#define WMMA_F16(A, B, C) \
    __builtin_amdgcn_wmma_f32_16x16x32_f16(false, (A), false, (B), (short)0, (C), false, false)

// ---- WMMA fragment loaders: all reads are contiguous 16B chunks -> ds_load_b128 ----

// A: 16x32 f16, row-major source with stride lda (halves). lda*2 must be 16B-multiple.
__device__ __forceinline__ v16h load_a_frag(const _Float16* base, int lda, int lane) {
    int m  = lane & 15;
    int kh = (lane >> 4) << 3;
    const _Float16* row = base + m * lda + kh;
    v8h lo = *(const v8h*)(row);
    v8h hi = *(const v8h*)(row + 16);
    return __builtin_shufflevector(lo, hi, 0, 1, 2, 3, 4, 5, 6, 7,
                                   8, 9, 10, 11, 12, 13, 14, 15);
}

// B fragment from a K-contiguous store: source row r holds the N-index.
// lanes 0-15: N=lane, K=0..7,16..23; lanes 16-31: K=8..15,24..31.
__device__ __forceinline__ v16h load_b_fragK(const _Float16* base, int ld, int lane) {
    int r = lane & 15;
    int o = (lane >> 4) << 3;
    const _Float16* row = base + r * ld + o;
    v8h lo = *(const v8h*)(row);
    v8h hi = *(const v8h*)(row + 16);
    return __builtin_shufflevector(lo, hi, 0, 1, 2, 3, 4, 5, 6, 7,
                                   8, 9, 10, 11, 12, 13, 14, 15);
}

// ---------------------------------------------------------------------------
// Kernel 0: convert + transpose weights f32 -> f16.
// ---------------------------------------------------------------------------
__global__ void cvt_weights(const float* __restrict__ wqkv, const float* __restrict__ wo,
                            _Float16* __restrict__ wqkv_t, _Float16* __restrict__ wo_t) {
    int i = blockIdx.x * blockDim.x + threadIdx.x;
    if (i < D_MODEL * QKV_COLS) {
        int k = i / QKV_COLS;
        int c = i - k * QKV_COLS;
        wqkv_t[c * D_MODEL + k] = (_Float16)wqkv[i];
    }
    if (i < D_MODEL * D_MODEL) {
        int k = i >> 8;
        int c = i & 255;
        wo_t[c * D_MODEL + k] = (_Float16)wo[i];
    }
}

// ---------------------------------------------------------------------------
// Kernel 1: QKV projection with double-buffered LDS tiles.
// Q,K -> [B,H,N,32] f16 ; V -> [B,H,32,N] f16 (transposed for attention B-frags)
// ---------------------------------------------------------------------------
#define XS_LD 40
#define WT_LD 40

__global__ __launch_bounds__(256) void qkv_gemm(
    const float* __restrict__ x, const _Float16* __restrict__ wt,
    const float* __restrict__ bqkv,
    _Float16* __restrict__ qh, _Float16* __restrict__ kh, _Float16* __restrict__ vh) {
    __shared__ __align__(16) _Float16 xs[2][64 * XS_LD];
    __shared__ __align__(16) _Float16 wsT[2][64 * WT_LD];   // [n][k]
    const int tid  = threadIdx.x;
    const int lane = tid & 31;
    const int wv   = tid >> 5;   // 0..7
    const int wr   = wv >> 1;    // row sub-tile 0..3 (16 rows each)
    const int wc   = wv & 1;     // col sub-tile 0..1 (32 cols each)
    const int m0   = blockIdx.x * 64;
    const int c0   = blockIdx.y * 64;

    const int srow = tid >> 2;
    const int scs  = (tid & 3) * 8;

    auto stage = [&](int buf, int k0) {
        const float* srcx = x + (size_t)(m0 + srow) * D_MODEL + k0 + scs;
#pragma unroll
        for (int i = 0; i < 8; ++i) xs[buf][srow * XS_LD + scs + i] = (_Float16)srcx[i];
        const _Float16* srcw = wt + (size_t)(c0 + srow) * D_MODEL + k0 + scs;
        *(v8h*)(&wsT[buf][srow * WT_LD + scs]) = *(const v8h*)srcw;
    };

    v8f acc0 = {};
    v8f acc1 = {};

    stage(0, 0);
    __syncthreads();
    for (int kk = 0; kk < D_MODEL / 32; ++kk) {
        int cur = kk & 1;
        if (kk + 1 < D_MODEL / 32) stage(cur ^ 1, (kk + 1) * 32);
        v16h a  = load_a_frag(&xs[cur][wr * 16 * XS_LD], XS_LD, lane);
        v16h b0 = load_b_fragK(&wsT[cur][(wc * 32 + 0) * WT_LD], WT_LD, lane);
        v16h b1 = load_b_fragK(&wsT[cur][(wc * 32 + 16) * WT_LD], WT_LD, lane);
        acc0 = WMMA_F16(a, b0, acc0);
        acc1 = WMMA_F16(a, b1, acc1);
        __syncthreads();
    }

    const int n_ = lane & 15;
    const int mb = (lane >> 4) * 8;
#pragma unroll
    for (int t = 0; t < 2; ++t) {
        const v8f* accp = t ? &acc1 : &acc0;
        int col   = c0 + wc * 32 + t * 16 + n_;
        float bia = bqkv[col];
        int mat = col >> 8;          // 0=Q 1=K 2=V
        int rem = col & 255;
        int h   = rem >> 5;
        int d   = rem & 31;
#pragma unroll
        for (int r = 0; r < 8; ++r) {
            int m    = m0 + wr * 16 + mb + r;
            int bidx = m >> 11;
            int nseq = m & (SEQ - 1);
            float v  = (*accp)[r] + bia;
            size_t bhh = (size_t)(bidx * N_HEADS + h);
            if (mat == 2) {
                vh[(bhh * D_K + d) * SEQ + nseq] = (_Float16)v;      // transposed
            } else {
                _Float16* dst = (mat == 0) ? qh : kh;
                dst[(bhh * SEQ + nseq) * D_K + d] = (_Float16)v;
            }
        }
    }
}

// ---------------------------------------------------------------------------
// Kernel 2: flash attention with distance bias; double-buffered K/V staging.
// Grid: (N/64, B*H). Block 128 thr (4 waves); wave owns 16 query rows.
// ---------------------------------------------------------------------------
#define KS_LD 40
#define VT_LD 72
#define PS_LD 72
#define NKB (SEQ / 64)

__global__ __launch_bounds__(128) void flash_attn(
    const _Float16* __restrict__ qh, const _Float16* __restrict__ kh,
    const _Float16* __restrict__ vh, const float* __restrict__ positions,
    const float* __restrict__ physics_bias, const float* __restrict__ distance_scale,
    _Float16* __restrict__ attn_out) {
    __shared__ __align__(16) _Float16 Ks[2][64 * KS_LD];    // [key][d]
    __shared__ __align__(16) _Float16 VsT[2][32 * VT_LD];   // [d][key]
    __shared__ __align__(16) _Float16 Ps[4 * 16 * PS_LD];
    __shared__ float pk_s[2][64];

    const int tid   = threadIdx.x;
    const int lane  = tid & 31;
    const int wv    = tid >> 5;          // 0..3
    const int bh    = blockIdx.y;        // 0..31
    const int b     = bh >> 3;
    const int h     = bh & 7;
    const int qrow0 = blockIdx.x * 64 + wv * 16;

    const float ds    = distance_scale[h];
    const float pb    = physics_bias[h];
    const float scale = 0.17677669529663687f;   // 1/sqrt(d_k)

    // Q fragment straight from global ([n][32] row-major, f16)
    v16h a_q = load_a_frag(qh + ((size_t)bh * SEQ + qrow0) * D_K, D_K, lane);

    const int n_ = lane & 15;
    const int mb = (lane >> 4) * 8;

    float pq[8];
#pragma unroll
    for (int r = 0; r < 8; ++r) pq[r] = positions[b * SEQ + qrow0 + mb + r];

    float m_i[8], l_i[8];
#pragma unroll
    for (int r = 0; r < 8; ++r) { m_i[r] = -1e30f; l_i[r] = 0.f; }
    v8f o0 = {};
    v8f o1 = {};

    _Float16* myP = Ps + wv * 16 * PS_LD;

    const int krow = tid >> 1;
    const int kcs  = (tid & 1) * 16;
    const int vrow = tid >> 2;          // d: 0..31
    const int vcs  = (tid & 3) * 16;    // key

    auto stage = [&](int buf, int key0) {
        const _Float16* ksrc = kh + ((size_t)bh * SEQ + key0 + krow) * D_K + kcs;
        *(v8h*)(&Ks[buf][krow * KS_LD + kcs])     = *(const v8h*)(ksrc);
        *(v8h*)(&Ks[buf][krow * KS_LD + kcs + 8]) = *(const v8h*)(ksrc + 8);
        const _Float16* vsrc = vh + ((size_t)bh * D_K + vrow) * SEQ + key0 + vcs;
        *(v8h*)(&VsT[buf][vrow * VT_LD + vcs])     = *(const v8h*)(vsrc);
        *(v8h*)(&VsT[buf][vrow * VT_LD + vcs + 8]) = *(const v8h*)(vsrc + 8);
        if (tid < 64) pk_s[buf][tid] = positions[b * SEQ + key0 + tid];
        if (key0 + 128 < SEQ) {          // L2 warm 2 blocks ahead
            __builtin_prefetch(ksrc + 128 * D_K, 0, 1);
            __builtin_prefetch(vsrc + 128, 0, 1);
        }
    };

    stage(0, 0);
    __syncthreads();

    for (int kb = 0; kb < NKB; ++kb) {
        const int cur = kb & 1;
        if (kb + 1 < NKB) stage(cur ^ 1, (kb + 1) * 64);   // overlaps compute below

        // S = Q K^T : 4 column tiles of 16 keys, contraction K=32=d_k
        v8f s[4];
#pragma unroll
        for (int ct = 0; ct < 4; ++ct) {
            v16h bk = load_b_fragK(&Ks[cur][ct * 16 * KS_LD], KS_LD, lane);
            v8f z = {};
            s[ct] = WMMA_F16(a_q, bk, z);
        }

        // fused physics bias + running row max
        float mnew[8];
#pragma unroll
        for (int r = 0; r < 8; ++r) mnew[r] = -1e30f;
#pragma unroll
        for (int ct = 0; ct < 4; ++ct) {
            float pk = pk_s[cur][ct * 16 + n_];
#pragma unroll
            for (int r = 0; r < 8; ++r) {
                float v = s[ct][r] * scale - ds * fabsf(pq[r] - pk) + pb;
                s[ct][r] = v;
                mnew[r] = fmaxf(mnew[r], v);
            }
        }
        // row reduce across the 16-lane half (xor 1/2/4/8 stays in-half)
#pragma unroll
        for (int off = 8; off >= 1; off >>= 1)
#pragma unroll
            for (int r = 0; r < 8; ++r)
                mnew[r] = fmaxf(mnew[r], __shfl_xor(mnew[r], off, 32));

        float alpha[8], lsum[8];
#pragma unroll
        for (int r = 0; r < 8; ++r) {
            float mn = fmaxf(m_i[r], mnew[r]);
            alpha[r] = __expf(m_i[r] - mn);
            m_i[r]   = mn;
            lsum[r]  = 0.f;
        }
        // P = exp(S - m) -> per-wave LDS (re-layout into A fragments)
#pragma unroll
        for (int ct = 0; ct < 4; ++ct) {
#pragma unroll
            for (int r = 0; r < 8; ++r) {
                float p = __expf(s[ct][r] - m_i[r]);
                lsum[r] += p;
                myP[(mb + r) * PS_LD + ct * 16 + n_] = (_Float16)p;
            }
        }
#pragma unroll
        for (int off = 8; off >= 1; off >>= 1)
#pragma unroll
            for (int r = 0; r < 8; ++r)
                lsum[r] += __shfl_xor(lsum[r], off, 32);
#pragma unroll
        for (int r = 0; r < 8; ++r) {
            l_i[r] = l_i[r] * alpha[r] + lsum[r];
            o0[r] *= alpha[r];
            o1[r] *= alpha[r];
        }
        // O += P V : A = P (16x32 keys), B from V^T rows (d = N index)
#pragma unroll
        for (int kt = 0; kt < 2; ++kt) {
            v16h ap  = load_a_frag(myP + kt * 32, PS_LD, lane);
            v16h bv0 = load_b_fragK(&VsT[cur][kt * 32], VT_LD, lane);              // d 0..15
            v16h bv1 = load_b_fragK(&VsT[cur][16 * VT_LD + kt * 32], VT_LD, lane); // d 16..31
            o0 = WMMA_F16(ap, bv0, o0);
            o1 = WMMA_F16(ap, bv1, o1);
        }
        __syncthreads();   // staging of next buffer + all reads of cur complete
    }

    // epilogue: attn_out[b, row, h*32 + d] (f16, [B,N,256])
#pragma unroll
    for (int r = 0; r < 8; ++r) {
        int row   = qrow0 + mb + r;
        float inv = 1.0f / l_i[r];
        size_t base = ((size_t)b * SEQ + row) * D_MODEL + h * D_K;
        attn_out[base + n_]      = (_Float16)(o0[r] * inv);
        attn_out[base + 16 + n_] = (_Float16)(o1[r] * inv);
    }
}

// ---------------------------------------------------------------------------
// Kernel 3: output projection. out = attn @ w_o + b_o (f32 out)
// ---------------------------------------------------------------------------
__global__ __launch_bounds__(256) void oproj_gemm(
    const _Float16* __restrict__ ain, const _Float16* __restrict__ wt,
    const float* __restrict__ bo, float* __restrict__ out) {
    __shared__ __align__(16) _Float16 xs[2][64 * XS_LD];
    __shared__ __align__(16) _Float16 wsT[2][64 * WT_LD];
    const int tid  = threadIdx.x;
    const int lane = tid & 31;
    const int wv   = tid >> 5;
    const int wr   = wv >> 1;
    const int wc   = wv & 1;
    const int m0   = blockIdx.x * 64;
    const int c0   = blockIdx.y * 64;

    const int srow = tid >> 2;
    const int scs  = (tid & 3) * 8;

    auto stage = [&](int buf, int k0) {
        const _Float16* srcx = ain + (size_t)(m0 + srow) * D_MODEL + k0 + scs;
        *(v8h*)(&xs[buf][srow * XS_LD + scs]) = *(const v8h*)srcx;
        const _Float16* srcw = wt + (size_t)(c0 + srow) * D_MODEL + k0 + scs;
        *(v8h*)(&wsT[buf][srow * WT_LD + scs]) = *(const v8h*)srcw;
    };

    v8f acc0 = {};
    v8f acc1 = {};

    stage(0, 0);
    __syncthreads();
    for (int kk = 0; kk < D_MODEL / 32; ++kk) {
        int cur = kk & 1;
        if (kk + 1 < D_MODEL / 32) stage(cur ^ 1, (kk + 1) * 32);
        v16h a  = load_a_frag(&xs[cur][wr * 16 * XS_LD], XS_LD, lane);
        v16h b0 = load_b_fragK(&wsT[cur][(wc * 32 + 0) * WT_LD], WT_LD, lane);
        v16h b1 = load_b_fragK(&wsT[cur][(wc * 32 + 16) * WT_LD], WT_LD, lane);
        acc0 = WMMA_F16(a, b0, acc0);
        acc1 = WMMA_F16(a, b1, acc1);
        __syncthreads();
    }

    const int n_ = lane & 15;
    const int mb = (lane >> 4) * 8;
#pragma unroll
    for (int t = 0; t < 2; ++t) {
        const v8f* accp = t ? &acc1 : &acc0;
        int col   = c0 + wc * 32 + t * 16 + n_;
        float bia = bo[col];
#pragma unroll
        for (int r = 0; r < 8; ++r) {
            int m = m0 + wr * 16 + mb + r;
            out[(size_t)m * D_MODEL + col] = (*accp)[r] + bia;
        }
    }
}

// ---------------------------------------------------------------------------
extern "C" void kernel_launch(void* const* d_in, const int* in_sizes, int n_in,
                              void* d_out, int out_size, void* d_ws, size_t ws_size,
                              hipStream_t stream) {
    const float* x              = (const float*)d_in[0];
    const float* positions      = (const float*)d_in[1];
    const float* w_qkv          = (const float*)d_in[2];
    const float* b_qkv          = (const float*)d_in[3];
    const float* w_o            = (const float*)d_in[4];
    const float* b_o            = (const float*)d_in[5];
    const float* physics_bias   = (const float*)d_in[6];
    const float* distance_scale = (const float*)d_in[7];
    float* out = (float*)d_out;

    char* ws = (char*)d_ws;
    size_t off = 0;
    auto alloc = [&](size_t bytes) -> void* {
        void* p = ws + off;
        off += (bytes + 255) & ~(size_t)255;
        return p;
    };
    const size_t HEAD_ELEMS = (size_t)BATCH * N_HEADS * SEQ * D_K;   // 2M
    _Float16* wqkv_t = (_Float16*)alloc((size_t)D_MODEL * QKV_COLS * 2);
    _Float16* wo_t   = (_Float16*)alloc((size_t)D_MODEL * D_MODEL * 2);
    _Float16* qh     = (_Float16*)alloc(HEAD_ELEMS * 2);
    _Float16* kh     = (_Float16*)alloc(HEAD_ELEMS * 2);
    _Float16* vh     = (_Float16*)alloc(HEAD_ELEMS * 2);   // [B,H,d,seq]
    _Float16* attn_h = (_Float16*)alloc((size_t)NROWS * D_MODEL * 2);
    (void)ws_size; (void)in_sizes; (void)n_in; (void)out_size;

    cvt_weights<<<dim3((D_MODEL * QKV_COLS + 255) / 256), dim3(256), 0, stream>>>(
        w_qkv, w_o, wqkv_t, wo_t);

    qkv_gemm<<<dim3(NROWS / 64, QKV_COLS / 64), dim3(256), 0, stream>>>(
        x, wqkv_t, b_qkv, qh, kh, vh);

    flash_attn<<<dim3(SEQ / 64, BATCH * N_HEADS), dim3(128), 0, stream>>>(
        qh, kh, vh, positions, physics_bias, distance_scale, attn_h);

    oproj_gemm<<<dim3(NROWS / 64, D_MODEL / 64), dim3(256), 0, stream>>>(
        attn_h, wo_t, b_o, out);
}
